// RegionLossV2_40973988004116
// MI455X (gfx1250) — compile-verified, matrix operand order checked by reading, not run on hardware
//
#include <hip/hip_runtime.h>
#include <hip/hip_bf16.h>
#include <math.h>

// ---- problem constants (from reference) ----
#define B_      64
#define C_      20
#define NA_     5
#define T_      50
#define H_      19
#define W_      19
#define HW_     361                  // H*W
#define NB_     (B_ * C_)            // 1280
#define A4_     (NA_ * HW_)          // 1805
#define TOTAL_  (NB_ * A4_)          // 2,310,400 cells (fits int32)
#define EPSV    1e-12f
#define BCE_E   1e-7f
#define THRESH_ 0.6f

typedef _Float16 v16h __attribute__((ext_vector_type(16)));
typedef float    v8f  __attribute__((ext_vector_type(8)));

// ---------------------------------------------------------------------------
// Wave-level sum via V_WMMA_F32_16X16X32_F16 with an all-ones B matrix.
// D = A x ones  => D[m][n] = rowsum_m(A).  Each lane deposits its value
// (Dekker-split into two f16 terms, same A row) into A; summing the 8 C/D
// VGPRs per lane yields sum(rows 0..7) on lanes 0-15 and sum(rows 8..15) on
// lanes 16-31; one shfl_xor(16) completes the full 32-lane fp32 sum.
// The total is invariant to the exact lane->(M,K) striping.
// ---------------------------------------------------------------------------
__device__ __forceinline__ float wave_sum_wmma(float v) {
  v16h a;
#pragma unroll
  for (int i = 0; i < 16; ++i) a[i] = (_Float16)0.0f;
  _Float16 h1 = (_Float16)v;
  float rem   = v - (float)h1;     // two-term split: h1 + h2 ~ v (fp32-accurate)
  a[0] = h1;
  a[1] = (_Float16)rem;
  v16h b;
#pragma unroll
  for (int i = 0; i < 16; ++i) b[i] = (_Float16)1.0f;
  v8f c = {};
  c = __builtin_amdgcn_wmma_f32_16x16x32_f16(false, a, false, b, (short)0, c,
                                             false, false);
  float s = ((c[0] + c[1]) + (c[2] + c[3])) + ((c[4] + c[5]) + (c[6] + c[7]));
  s += __shfl_xor(s, 16, 32);      // combine the two half-wave row groups
  return s;                        // full sum broadcast to all 32 lanes
}

__device__ __forceinline__ float iou_xywh(float x1, float y1, float w1, float h1,
                                          float x2, float y2, float w2, float h2) {
  float l = fmaxf(x1 - w1 * 0.5f, x2 - w2 * 0.5f);
  float r = fminf(x1 + w1 * 0.5f, x2 + w2 * 0.5f);
  float t = fmaxf(y1 - h1 * 0.5f, y2 - h2 * 0.5f);
  float b = fminf(y1 + h1 * 0.5f, y2 + h2 * 0.5f);
  float inter = fmaxf(r - l, 0.0f) * fmaxf(b - t, 0.0f);
  float uni   = w1 * h1 + w2 * h2 - inter;
  return inter / fmaxf(uni, EPSV);
}

__device__ __forceinline__ float bce_term(float p, float t) {
  p = fminf(fmaxf(p, BCE_E), 1.0f - BCE_E);
  return -(t * logf(p) + (1.0f - t) * logf(1.0f - p));
}

__device__ __forceinline__ float sigmoidf(float x) {
  return 1.0f / (1.0f + expf(-x));
}

// acc[0]=xy BCE sum, acc[1]=w^2+h^2+conf^2 sum, acc[2]=nll sum, acc[3]=nsel
__global__ void k0_zero(float* acc) {
  if (threadIdx.x < 4) acc[threadIdx.x] = 0.0f;
}

// ---------------------------------------------------------------------------
// K1: per NB-row target prep (valid cumprod, best anchor, tw/th, packed meta)
// meta = valid ? (bn | gi<<3 | gj<<8 | cls<<13) : -1
// ---------------------------------------------------------------------------
__global__ void k1_targets(const float* __restrict__ target,
                           const float* __restrict__ anchors,
                           float* __restrict__ tgx, float* __restrict__ tgy,
                           float* __restrict__ tgw, float* __restrict__ tgh,
                           float* __restrict__ ttw, float* __restrict__ tth,
                           int* __restrict__ tmeta) {
  int nb = blockIdx.x * blockDim.x + threadIdx.x;
  if (nb >= NB_) return;
  const float* tp = target + (long)nb * (T_ * 5);
  int vflag = 1;
  for (int t = 0; t < T_; ++t) {
    float c0  = tp[t * 5 + 0];
    float x01 = tp[t * 5 + 1];
    float y01 = tp[t * 5 + 2];
    float w01 = tp[t * 5 + 3];
    float h01 = tp[t * 5 + 4];
    if (x01 == 0.0f) vflag = 0;            // inclusive cumprod semantics
    float gx = x01 * (float)W_, gy = y01 * (float)H_;
    float gw = w01 * (float)W_, gh = h01 * (float)H_;
    int bn = 0; float best = -1.0f;
    for (int a = 0; a < NA_; ++a) {
      float aw = anchors[2 * a], ah = anchors[2 * a + 1];
      float inter = fminf(gw, aw) * fminf(gh, ah);
      float uni   = gw * gh + aw * ah - inter;
      float ratio = inter / fmaxf(uni, EPSV);
      if (ratio > best) { best = ratio; bn = a; }   // first max wins (strict >)
    }
    int gi = (int)gx, gj = (int)gy;
    int cls = (int)c0;
    int idx = nb * T_ + t;
    tgx[idx] = gx; tgy[idx] = gy; tgw[idx] = gw; tgh[idx] = gh;
    ttw[idx] = logf(fmaxf(gw, EPSV) / anchors[2 * bn]);
    tth[idx] = logf(fmaxf(gh, EPSV) / anchors[2 * bn + 1]);
    tmeta[idx] = vflag ? (bn | (gi << 3) | (gj << 8) | (cls << 13)) : -1;
  }
}

// ---------------------------------------------------------------------------
// K2: per-cell pass. coord_mask == 1 everywhere in the reference, so loss_x/y
// are dense BCE vs (tx,ty) with 0.5 defaults; loss_w/h dense MSE vs 0 default.
// Scatter is turned into a gather: the *last* valid target with
// (best_n==n, gi==i, gj==j) wins (forward loop keeps overwriting).
// ---------------------------------------------------------------------------
__global__ void __launch_bounds__(256)
k2_cells(const float* __restrict__ out, const float* __restrict__ anchors,
         const float* __restrict__ tgx, const float* __restrict__ tgy,
         const float* __restrict__ tgw, const float* __restrict__ tgh,
         const float* __restrict__ ttw, const float* __restrict__ tth,
         const int* __restrict__ tmeta, signed char* __restrict__ mcls,
         float* __restrict__ acc) {
  int idx = blockIdx.x * blockDim.x + threadIdx.x;
  bool active = idx < TOTAL_;
  int i0 = active ? idx : 0;

  int nb = i0 / A4_;
  int r  = i0 - nb * A4_;
  int n  = r / HW_;
  int rr = r - n * HW_;
  int jj = rr / W_;
  int ii = rr - jj * W_;

  long base = ((long)nb * 30 + n * 6) * HW_ + rr;
  float xl = out[base];
  float yl = out[base + HW_];
  float wv = out[base + 2 * HW_];
  float hv = out[base + 3 * HW_];
  float cl = out[base + 4 * HW_];

  float sx = sigmoidf(xl), sy = sigmoidf(yl), sconf = sigmoidf(cl);
  float aw = anchors[2 * n], ah = anchors[2 * n + 1];
  float px = sx + (float)ii;
  float py = sy + (float)jj;
  float pw = expf(wv) * aw;
  float ph = expf(hv) * ah;

  float cur_iou = 0.0f;
  int   mt = -1;
  const int tb = nb * T_;
  for (int t = 0; t < T_; ++t) {
    int meta = tmeta[tb + t];
    if (meta < 0) continue;                 // invalid target
    float gx = tgx[tb + t], gy = tgy[tb + t];
    float gw = tgw[tb + t], gh = tgh[tb + t];
    cur_iou = fmaxf(cur_iou, iou_xywh(px, py, pw, ph, gx, gy, gw, gh));
    int bn = meta & 7, gi = (meta >> 3) & 31, gj = (meta >> 8) & 31;
    if (bn == n && gi == ii && gj == jj) mt = t;   // last valid match wins
  }

  float tx = 0.5f, ty = 0.5f, tw = 0.0f, th = 0.0f, tconf = 0.0f, cfm;
  int mclass = -1;
  if (mt >= 0) {
    int meta = tmeta[tb + mt];
    int gi = (meta >> 3) & 31, gj = (meta >> 8) & 31;
    mclass = (meta >> 13) & 255;
    float gx = tgx[tb + mt], gy = tgy[tb + mt];
    float gw = tgw[tb + mt], gh = tgh[tb + mt];
    tx = gx - (float)gi;
    ty = gy - (float)gj;
    tw = ttw[tb + mt];
    th = tth[tb + mt];
    tconf = iou_xywh(gx, gy, gw, gh, px, py, pw, ph);
    cfm = 1.0f;
  } else {
    cfm = (cur_iou > THRESH_) ? 0.0f : 1.0f;
  }

  float c_xy  = bce_term(sx, tx) + bce_term(sy, ty);
  float dw = wv - tw, dh = hv - th, dc = (sconf - tconf) * cfm;
  float c_whc = dw * dw + dh * dh + dc * dc;
  if (!active) { c_xy = 0.0f; c_whc = 0.0f; }
  if (active) mcls[i0] = (signed char)mclass;

  // wave-level WMMA reduction, then block combine, then 2 atomics per block
  __shared__ float red0[8], red1[8];
  int wave = threadIdx.x >> 5, lane = threadIdx.x & 31;
  float s0 = wave_sum_wmma(c_xy);
  float s1 = wave_sum_wmma(c_whc);
  if (lane == 0) { red0[wave] = s0; red1[wave] = s1; }
  __syncthreads();
  if (threadIdx.x == 0) {
    float t0 = 0.0f, t1 = 0.0f;
    for (int w = 0; w < 8; ++w) { t0 += red0[w]; t1 += red1[w]; }
    atomicAdd(&acc[0], t0);
    atomicAdd(&acc[1], t1);
  }
}

// ---------------------------------------------------------------------------
// K3: class loss. For each (b, a4) cell: count matches across the C replica
// dim; if exactly one, 20-way log-softmax NLL on the cls-logit column.
// ---------------------------------------------------------------------------
__global__ void __launch_bounds__(256)
k3_cls(const float* __restrict__ out, const signed char* __restrict__ mcls,
       float* __restrict__ acc) {
  int s = blockIdx.x * blockDim.x + threadIdx.x;
  bool active = s < (B_ * A4_);
  int s0 = active ? s : 0;
  int b  = s0 / A4_;
  int a4 = s0 - b * A4_;
  int n  = a4 / HW_;
  int rr = a4 - n * HW_;

  int count = 0, csum = 0;
  for (int c = 0; c < C_; ++c) {
    int m = (int)mcls[(long)(b * C_ + c) * A4_ + a4];
    if (m >= 0) { ++count; csum += m; }
  }

  float nll = 0.0f, selc = 0.0f;
  if (active && count == 1) {
    float lg[C_];
    float mx = -INFINITY;
    for (int c = 0; c < C_; ++c) {
      lg[c] = out[((long)(b * C_ + c) * 30 + n * 6 + 5) * HW_ + rr];
      mx = fmaxf(mx, lg[c]);
    }
    float se = 0.0f;
    for (int c = 0; c < C_; ++c) se += expf(lg[c] - mx);
    float lse = mx + logf(se);
    nll  = lse - lg[csum];   // csum == the single matched class id
    selc = 1.0f;
  }

  __shared__ float red0[8], red1[8];
  int wave = threadIdx.x >> 5, lane = threadIdx.x & 31;
  float s0r = wave_sum_wmma(nll);
  float s1r = wave_sum_wmma(selc);
  if (lane == 0) { red0[wave] = s0r; red1[wave] = s1r; }
  __syncthreads();
  if (threadIdx.x == 0) {
    float t0 = 0.0f, t1 = 0.0f;
    for (int w = 0; w < 8; ++w) { t0 += red0[w]; t1 += red1[w]; }
    atomicAdd(&acc[2], t0);
    atomicAdd(&acc[3], t1);
  }
}

__global__ void k4_final(const float* __restrict__ acc, float* __restrict__ outp) {
  if (blockIdx.x == 0 && threadIdx.x == 0) {
    const float N = (float)TOTAL_;
    float nsel = fmaxf(acc[3], 1.0f);
    outp[0] = acc[0] / N + 0.5f * acc[1] / N + acc[2] / nsel;
  }
}

// ---------------------------------------------------------------------------
extern "C" void kernel_launch(void* const* d_in, const int* in_sizes, int n_in,
                              void* d_out, int out_size, void* d_ws, size_t ws_size,
                              hipStream_t stream) {
  (void)in_sizes; (void)n_in; (void)out_size; (void)ws_size;
  const float* out_p   = (const float*)d_in[0];
  const float* target  = (const float*)d_in[1];
  const float* anchors = (const float*)d_in[2];

  char* ws = (char*)d_ws;
  float* acc = (float*)ws;                         // 4 floats (64B reserved)
  float* tgx = (float*)(ws + 64);                  // NB_*T_ = 64000 each
  float* tgy = tgx + NB_ * T_;
  float* tgw = tgy + NB_ * T_;
  float* tgh = tgw + NB_ * T_;
  float* ttw = tgh + NB_ * T_;
  float* tth = ttw + NB_ * T_;
  int*   tmeta = (int*)(tth + NB_ * T_);
  signed char* mcls = (signed char*)(tmeta + NB_ * T_);   // TOTAL_ bytes

  k0_zero<<<1, 64, 0, stream>>>(acc);
  k1_targets<<<(NB_ + 255) / 256, 256, 0, stream>>>(target, anchors, tgx, tgy,
                                                    tgw, tgh, ttw, tth, tmeta);
  k2_cells<<<(TOTAL_ + 255) / 256, 256, 0, stream>>>(out_p, anchors, tgx, tgy,
                                                     tgw, tgh, ttw, tth, tmeta,
                                                     mcls, acc);
  k3_cls<<<(B_ * A4_ + 255) / 256, 256, 0, stream>>>(out_p, mcls, acc);
  k4_final<<<1, 32, 0, stream>>>(acc, (float*)d_out);
}
